// ContrastiveLoss_44779329028211
// MI455X (gfx1250) — matrix operand order repaired and checked by reading, not compiled, not run on hardware
//
#include <hip/hip_runtime.h>

// ---------------------------------------------------------------------------
// ContrastiveLoss on MI455X (gfx1250, wave32, WMMA + async global->LDS)
//
//   loss = mean_i [ C + log(sum_j exp(dot_ij/T - C)) - dot_ii/T ],  C = 1/T
//
// K1: L2-normalize rows, split fp32 -> bf16 hi/lo, zero accumulators.
// K2: tiled Gram via V_WMMA_F32_16X16X32_BF16 (hi*hi + hi*lo + lo*hi),
//     fused exp + row-sums (16384^2 logits never materialized), diagonal
//     pulled from the C fragment.  B panels double-buffered in LDS via
//     GLOBAL_LOAD_ASYNC_TO_LDS_B128 (ASYNCcnt) so L2 latency hides under
//     the 24 WMMAs of the current tile; one barrier per tile.
// K3: finalize per-row terms, reduce to scalar.
// ---------------------------------------------------------------------------

#define N_ROWS 16384
#define DIM    256
#define INV_T  14.285714285714286f   // 1 / 0.07

typedef __bf16 bf16;
typedef __attribute__((ext_vector_type(16))) __bf16 v16bf;
typedef __attribute__((ext_vector_type(8)))  __bf16 v8bf;
typedef __attribute__((ext_vector_type(4)))  __bf16 v4bf;
typedef __attribute__((ext_vector_type(8)))  float  v8f;

// --- gfx1250 async global->LDS copy (ASYNCcnt), 16 bytes per lane ----------
// LDS flat aperture keeps the LDS byte offset in addr[31:0] (ISA 10.2), so
// truncating the generic pointer yields the DS address the instruction wants.
__device__ __forceinline__ void async_copy_b128(void* lds_dst, const void* gsrc) {
  unsigned           ldsa = (unsigned)(uintptr_t)lds_dst;
  unsigned long long ga   = (unsigned long long)(uintptr_t)gsrc;
  asm volatile("global_load_async_to_lds_b128 %0, %1, off"
               :: "v"(ldsa), "v"(ga)
               : "memory");
}
__device__ __forceinline__ void wait_async0() {
  asm volatile("s_wait_asynccnt 0x0" ::: "memory");
}

// ---------------------------------------------------------------------------
// Kernel 1: normalize + hi/lo bf16 split.  One wave per row (8 rows / block).
// ---------------------------------------------------------------------------
__global__ __launch_bounds__(256) void k_normalize_split(
    const float* __restrict__ X, bf16* __restrict__ Ehi, bf16* __restrict__ Elo,
    float* __restrict__ rowS, float* __restrict__ out)
{
  const int tid  = threadIdx.x;
  const int lane = tid & 31;
  const int wave = tid >> 5;
  const int row  = blockIdx.x * 8 + wave;
  const int gid  = blockIdx.x * 256 + tid;

  if (gid < N_ROWS) rowS[gid] = 0.0f;   // clear accumulator every launch
  if (gid == 0)     out[0]    = 0.0f;

  const float4* xr = (const float4*)(X + (size_t)row * DIM);
  float4 x0 = xr[lane];        // elements lane*4 .. +3
  float4 x1 = xr[lane + 32];   // elements 128 + lane*4 .. +3

  float ss = x0.x*x0.x + x0.y*x0.y + x0.z*x0.z + x0.w*x0.w
           + x1.x*x1.x + x1.y*x1.y + x1.z*x1.z + x1.w*x1.w;
  ss += __shfl_xor(ss, 1, 32);
  ss += __shfl_xor(ss, 2, 32);
  ss += __shfl_xor(ss, 4, 32);
  ss += __shfl_xor(ss, 8, 32);
  ss += __shfl_xor(ss, 16, 32);

  float nrm = fmaxf(sqrtf(ss), 1e-12f);
  float inv = 1.0f / nrm;

  float v[8] = { x0.x*inv, x0.y*inv, x0.z*inv, x0.w*inv,
                 x1.x*inv, x1.y*inv, x1.z*inv, x1.w*inv };

  v4bf h0, h1, l0, l1;
#pragma unroll
  for (int i = 0; i < 4; ++i) {
    __bf16 a = (__bf16)v[i];
    h0[i] = a;  l0[i] = (__bf16)(v[i]     - (float)a);
    __bf16 b = (__bf16)v[i + 4];
    h1[i] = b;  l1[i] = (__bf16)(v[i + 4] - (float)b);
  }
  v4bf* dh = (v4bf*)(Ehi + (size_t)row * DIM);
  v4bf* dl = (v4bf*)(Elo + (size_t)row * DIM);
  dh[lane]      = h0;
  dh[lane + 32] = h1;
  dl[lane]      = l0;
  dl[lane + 32] = l1;
}

// ---------------------------------------------------------------------------
// Kernel 2: Gram + fused exp row-sums, async double-buffered B panels.
//   grid = (128 row-blocks of 128 rows, 4 column splits), block = 256 (8 waves)
// ---------------------------------------------------------------------------
__global__ __launch_bounds__(256) void k_gram_lse(
    const bf16* __restrict__ Ehi, const bf16* __restrict__ Elo,
    float* __restrict__ rowS, float* __restrict__ diag)
{
  const int tid      = threadIdx.x;
  const int lane     = tid & 31;
  const int wave     = tid >> 5;                 // 0..7
  const int row_base = blockIdx.x * 128 + wave * 16;
  const int ct_begin = blockIdx.y * 256;         // 16-wide column tiles
  const int ct_end   = ct_begin + 256;

  // two buffers x (hi 8KB + lo 8KB) = 32 KB
  __shared__ __attribute__((aligned(32))) bf16 panel_hi[2][16 * DIM];
  __shared__ __attribute__((aligned(32))) bf16 panel_lo[2][16 * DIM];

  const int m = lane & 15;      // A: row within tile / B: col within tile
  const int h = lane >> 4;      // lane half (K interleave select)

  // ---- resident A fragments (A 16x32, lane row = m, half h -> 16 contiguous K) ----------------
  v16bf ahi[8], alo[8];
  {
    const bf16* rh = Ehi + (size_t)(row_base + m) * DIM;
    const bf16* rl = Elo + (size_t)(row_base + m) * DIM;
#pragma unroll
    for (int kt = 0; kt < 8; ++kt) {
      v8bf a0 = *(const v8bf*)(rh + kt * 32 + h * 16);
      v8bf a1 = *(const v8bf*)(rh + kt * 32 + h * 16 + 8);
      ahi[kt] = __builtin_shufflevector(a0, a1, 0,1,2,3,4,5,6,7,8,9,10,11,12,13,14,15);
      v8bf b0 = *(const v8bf*)(rl + kt * 32 + h * 16);
      v8bf b1 = *(const v8bf*)(rl + kt * 32 + h * 16 + 8);
      alo[kt] = __builtin_shufflevector(b0, b1, 0,1,2,3,4,5,6,7,8,9,10,11,12,13,14,15);
    }
  }

  // stage one 16x256 hi/lo panel (4x b128 per thread) into buffer `buf`
  auto stage = [&](int buf, int ct) {
    const uint4* sh = (const uint4*)(Ehi + (size_t)ct * 16 * DIM);
    const uint4* sl = (const uint4*)(Elo + (size_t)ct * 16 * DIM);
    async_copy_b128((uint4*)panel_hi[buf] + tid,       sh + tid);
    async_copy_b128((uint4*)panel_hi[buf] + tid + 256, sh + tid + 256);
    async_copy_b128((uint4*)panel_lo[buf] + tid,       sl + tid);
    async_copy_b128((uint4*)panel_lo[buf] + tid + 256, sl + tid + 256);
  };

  v8f   rs     = {};                       // running exp row-sums (C layout)
  float mydiag = 0.0f;
  const int diag_ct = row_base >> 4;       // column tile holding the diagonal

  // prologue: fill buffer 0
  stage(0, ct_begin);
  wait_async0();
  __syncthreads();

  for (int ct = ct_begin; ct < ct_end; ++ct) {
    const int cur = (ct - ct_begin) & 1;
    if (ct + 1 < ct_end) stage(cur ^ 1, ct + 1);   // overlap next panel fetch

    v8f c = {};
#pragma unroll
    for (int kt = 0; kt < 8; ++kt) {
      // B 32x16 operand (same K order as A): lane col = m, half h -> 16 contiguous K
      const bf16* bph = panel_hi[cur] + (size_t)m * DIM + kt * 32 + h * 16;
      const bf16* bpl = panel_lo[cur] + (size_t)m * DIM + kt * 32 + h * 16;
      v16bf bhi = *(const v16bf*)bph;
      v16bf blo = *(const v16bf*)bpl;
      c = __builtin_amdgcn_wmma_f32_16x16x32_bf16(false, ahi[kt], false, bhi,
                                                  (short)0, c, false, false);
      c = __builtin_amdgcn_wmma_f32_16x16x32_bf16(false, ahi[kt], false, blo,
                                                  (short)0, c, false, false);
      c = __builtin_amdgcn_wmma_f32_16x16x32_bf16(false, alo[kt], false, bhi,
                                                  (short)0, c, false, false);
    }

    if (ct == diag_ct) {                   // pull dot_ii out of the fragment
#pragma unroll
      for (int v = 0; v < 8; ++v) {        // C layout: M = v + 8*h, N = lane&15
        int want = (lane < 16) ? v : (v + 24);
        if (lane == want) mydiag = c[v];
      }
    }

#pragma unroll
    for (int v = 0; v < 8; ++v)            // exp(logit - C) with C = 1/T
      rs[v] += __expf((c[v] - 1.0f) * INV_T);

    wait_async0();       // own async writes into the other buffer complete
    __syncthreads();     // all waves done with `cur`, next buffer visible
  }

  // ---- per-row reduction: sum the 16 N-lanes inside each half ------------
#pragma unroll
  for (int v = 0; v < 8; ++v) {
    rs[v] += __shfl_xor(rs[v], 1, 32);
    rs[v] += __shfl_xor(rs[v], 2, 32);
    rs[v] += __shfl_xor(rs[v], 4, 32);
    rs[v] += __shfl_xor(rs[v], 8, 32);
  }
  if ((lane & 15) == 0) {                  // lane 0 -> rows 0..7, lane 16 -> 8..15
#pragma unroll
    for (int v = 0; v < 8; ++v)
      atomicAdd(&rowS[row_base + h * 8 + v], rs[v]);
  }
  if (diag_ct >= ct_begin && diag_ct < ct_end) {   // owning split writes diag
    if (lane < 8)        diag[row_base + lane]            = mydiag;
    else if (lane >= 24) diag[row_base + 8 + (lane - 24)] = mydiag;
  }
}

// ---------------------------------------------------------------------------
// Kernel 3: loss_i = C + log(S_i) - dot_ii/T ; mean over rows -> scalar.
// ---------------------------------------------------------------------------
__global__ __launch_bounds__(256) void k_finalize(
    const float* __restrict__ rowS, const float* __restrict__ diag,
    float* __restrict__ out)
{
  int i = blockIdx.x * 256 + threadIdx.x;
  float term = 0.0f;
  if (i < N_ROWS)
    term = INV_T + __logf(rowS[i]) - diag[i] * INV_T;
  term += __shfl_xor(term, 1, 32);
  term += __shfl_xor(term, 2, 32);
  term += __shfl_xor(term, 4, 32);
  term += __shfl_xor(term, 8, 32);
  term += __shfl_xor(term, 16, 32);
  if ((threadIdx.x & 31) == 0)
    atomicAdd(out, term * (1.0f / (float)N_ROWS));
}

// ---------------------------------------------------------------------------
extern "C" void kernel_launch(void* const* d_in, const int* in_sizes, int n_in,
                              void* d_out, int out_size, void* d_ws, size_t ws_size,
                              hipStream_t stream) {
  (void)in_sizes; (void)n_in; (void)out_size; (void)ws_size;
  const float* X = (const float*)d_in[0];
  float* out = (float*)d_out;

  bf16*  Ehi  = (bf16*)d_ws;                                   // 8 MB
  bf16*  Elo  = Ehi + (size_t)N_ROWS * DIM;                    // 8 MB
  float* rowS = (float*)(Elo + (size_t)N_ROWS * DIM);          // 64 KB
  float* diag = rowS + N_ROWS;                                 // 64 KB

  k_normalize_split<<<N_ROWS / 8, 256, 0, stream>>>(X, Ehi, Elo, rowS, out);

  dim3 grid(N_ROWS / 128, 4);   // 128 row-blocks x 4 column splits
  k_gram_lse<<<grid, 256, 0, stream>>>(Ehi, Elo, rowS, diag);

  k_finalize<<<(N_ROWS + 255) / 256, 256, 0, stream>>>(rowS, diag, out);
}
